// Glm4MoeDecoderLayer_85255100825930
// MI455X (gfx1250) — compile-verified, matrix operand order checked by reading, not run on hardware
//
#include <hip/hip_runtime.h>
#include <hip/hip_bf16.h>
#include <math.h>

// ---------------- problem constants (from reference) ----------------
constexpr int Bc  = 2;
constexpr int Sc  = 1024;
constexpr int Hc  = 2048;
constexpr int NHc = 16;
constexpr int NKVc= 4;
constexpr int Dc  = 128;
constexpr int Ec  = 16;
constexpr int Kc  = 2;
constexpr int Fc  = 1408;
constexpr int FSc = 1408;
constexpr int Tc  = Bc * Sc;            // 2048 tokens
constexpr float EPSc = 1e-5f;
constexpr float RSFc = 1.0f;

// ---------------- WMMA types ----------------
typedef __attribute__((ext_vector_type(16))) __bf16 v16bf;
typedef __attribute__((ext_vector_type(8)))  float  v8f;

union FragU { uint4 q[2]; v16bf v; };

__device__ __forceinline__ unsigned short f2bf(float f) {
    unsigned int u = __float_as_uint(f);
    unsigned int r = u + 0x7FFFu + ((u >> 16) & 1u);   // round-to-nearest-even
    return (unsigned short)(r >> 16);
}

__device__ __forceinline__ uint2 f4tobf(float4 v) {
    uint2 r;
    r.x = (unsigned)f2bf(v.x) | ((unsigned)f2bf(v.y) << 16);
    r.y = (unsigned)f2bf(v.z) | ((unsigned)f2bf(v.w) << 16);
    return r;
}

// =====================================================================
// Generic bf16 WMMA GEMM:  C[z] = A[z] @ B[z] (+ Cadd)
//   - tile: BM=128 x BN=128, BK=32; 256 threads = 8 waves;
//     wave tile 32x64 -> acc[2][4] -> 8 v_wmma per K step
//   - z offsets: base + (z/zdiv)*s1 + (z%zdiv)*s2  (covers [b][h] batching)
//   - transB: B element (k,n) at B[n*ldb+k] (used for Q @ K^T)
//   - cnt/off: dynamic per-z row count/offset (expert GEMMs)
//   - gather: indirection for A rows (token gather for expert gate_up)
// Preconditions exploited: Kd % 32 == 0; lda/ldb multiples of 4.
// Row pointers are K-invariant and hoisted out of the K loop.
// =====================================================================
#define BM 128
#define BN 128
#define BK 32

__global__ __launch_bounds__(256)
void gemm_bf16(const float* __restrict__ A, const float* __restrict__ Bm,
               const float* __restrict__ Cadd, float* __restrict__ C,
               int M, int N, int Kd,
               long lda, long ldb, long ldc,
               long sA1, long sA2, long sB1, long sB2, long sC1, long sC2,
               int zdiv, int transB,
               const int* __restrict__ cnt, const int* __restrict__ off,
               const int* __restrict__ gather)
{
    __shared__ __align__(16) unsigned short sA[BM * BK];        // [m][k]
    __shared__ __align__(16) unsigned short sB[BN * BK];        // [n][k] (transposed stage)

    const int z = blockIdx.z;
    const float* Az = A  + (long)(z / zdiv) * sA1 + (long)(z % zdiv) * sA2;
    const float* Bz = Bm + (long)(z / zdiv) * sB1 + (long)(z % zdiv) * sB2;
    float*       Cz = C  + (long)(z / zdiv) * sC1 + (long)(z % zdiv) * sC2;
    const float* Dz = Cadd ? (Cadd + (long)(z / zdiv) * sC1 + (long)(z % zdiv) * sC2) : nullptr;

    int rowbase = 0;
    if (cnt) { M = cnt[z]; rowbase = off[z]; }

    const int m0 = blockIdx.y * BM;
    const int n0 = blockIdx.x * BN;
    if (m0 >= M) return;

    const int tid  = threadIdx.x;
    const int lane = tid & 31;
    const int wave = tid >> 5;
    const int wm   = (wave & 3) * 32;      // wave M offset in block tile
    const int wn   = (wave >> 2) * 64;     // wave N offset in block tile
    const int half = lane >> 4;
    const int lr   = lane & 15;

    const bool fullM = (m0 + BM <= M);
    const bool fullN = (n0 + BN <= N);

    // ---------- hoist K-invariant staging addresses ----------
    // A: 4096 elems = 4 chunks/thread of float4
    const float* aPtr[4]; bool aOK[4]; int aOff[4];
    #pragma unroll
    for (int i = 0; i < 4; ++i) {
        int idx = tid + i * 256;        // 0..1023
        int r   = idx >> 3;             // row 0..127
        int c   = (idx & 7) * 4;        // k offset 0,4,...,28
        int gm  = m0 + r;
        bool ok = fullM || (gm < M);
        long arow = 0;
        if (ok) arow = gather ? (long)gather[rowbase + gm] : (long)(rowbase + gm);
        aOK[i]  = ok;
        aPtr[i] = Az + arow * lda + c;
        aOff[i] = r * BK + c;
    }
    // B: 4096 elems = 4 chunks/thread of float4 (layout depends on transB)
    const float* bPtr[4]; bool bOK[4]; int bOff[4]; int bN0[4];
    #pragma unroll
    for (int i = 0; i < 4; ++i) {
        int idx = tid + i * 256;        // 0..1023
        if (transB) {
            int nn = idx >> 3;          // 0..127
            int cc = (idx & 7) * 4;     // k offset
            int gn = n0 + nn;
            bOK[i]  = fullN || (gn < N);
            bPtr[i] = Bz + (long)(bOK[i] ? gn : 0) * ldb + cc;
            bOff[i] = nn * BK + cc;
            bN0[i]  = gn;
        } else {
            int kk = idx >> 5;          // 0..31
            int nc = (idx & 31) * 4;    // 0,4,...,124
            bOK[i]  = fullN || (n0 + nc + 3 < N);
            bPtr[i] = Bz + (long)kk * ldb + (n0 + nc);
            bOff[i] = nc * BK + kk;     // first of 4 scattered rows
            bN0[i]  = n0 + nc;
        }
    }

    v8f acc[2][4];
    #pragma unroll
    for (int i = 0; i < 2; ++i)
        #pragma unroll
        for (int j = 0; j < 4; ++j)
            #pragma unroll
            for (int e = 0; e < 8; ++e) acc[i][j][e] = 0.0f;

    for (int k0 = 0; k0 < Kd; k0 += BK) {
        const bool more = (k0 + BK) < Kd;
        // ---- stage A tile: b128 loads -> packed bf16 ds_store_b64 ----
        #pragma unroll
        for (int i = 0; i < 4; ++i) {
            float4 v = make_float4(0.f, 0.f, 0.f, 0.f);
            if (aOK[i]) {
                const float* p = aPtr[i] + k0;
                v = *(const float4*)p;
                if (more) __builtin_prefetch(p + BK, 0, 0);
            }
            *(uint2*)(sA + aOff[i]) = f4tobf(v);
        }
        // ---- stage B tile into n-major LDS ----
        if (transB) {
            #pragma unroll
            for (int i = 0; i < 4; ++i) {
                float4 v = make_float4(0.f, 0.f, 0.f, 0.f);
                if (bOK[i]) {
                    const float* p = bPtr[i] + k0;
                    v = *(const float4*)p;
                    if (more) __builtin_prefetch(p + BK, 0, 0);
                }
                *(uint2*)(sB + bOff[i]) = f4tobf(v);
            }
        } else {
            #pragma unroll
            for (int i = 0; i < 4; ++i) {
                const float* p = bPtr[i] + (long)k0 * ldb;
                float4 v = make_float4(0.f, 0.f, 0.f, 0.f);
                if (bOK[i]) {
                    v = *(const float4*)p;
                } else {
                    if (bN0[i] + 0 < N) v.x = p[0];
                    if (bN0[i] + 1 < N) v.y = p[1];
                    if (bN0[i] + 2 < N) v.z = p[2];
                    if (bN0[i] + 3 < N) v.w = p[3];
                }
                if (more) __builtin_prefetch(p + (long)BK * ldb, 0, 0);
                sB[bOff[i] + 0 * BK] = f2bf(v.x);
                sB[bOff[i] + 1 * BK] = f2bf(v.y);
                sB[bOff[i] + 2 * BK] = f2bf(v.z);
                sB[bOff[i] + 3 * BK] = f2bf(v.w);
            }
        }
        __syncthreads();

        // ---- fragment loads per documented 16-bit A layout ----
        // lane<16: M=lane, K in {0..7, 16..23}; lane>=16: M=lane-16, K in {8..15, 24..31}
        FragU a[2], b[4];
        #pragma unroll
        for (int mi = 0; mi < 2; ++mi) {
            const uint4* p = reinterpret_cast<const uint4*>(sA + (wm + mi * 16 + lr) * BK);
            a[mi].q[0] = p[half];
            a[mi].q[1] = p[half + 2];
        }
        #pragma unroll
        for (int ni = 0; ni < 4; ++ni) {
            const uint4* p = reinterpret_cast<const uint4*>(sB + (wn + ni * 16 + lr) * BK);
            b[ni].q[0] = p[half];
            b[ni].q[1] = p[half + 2];
        }

        #pragma unroll
        for (int mi = 0; mi < 2; ++mi)
            #pragma unroll
            for (int ni = 0; ni < 4; ++ni)
                acc[mi][ni] = __builtin_amdgcn_wmma_f32_16x16x32_bf16(
                    false, a[mi].v, false, b[ni].v, (short)0, acc[mi][ni], false, false);

        __syncthreads();
    }

    // ---- epilogue: C layout VGPR r -> (M = r + 8*half, N = lane&15) ----
    #pragma unroll
    for (int mi = 0; mi < 2; ++mi)
        #pragma unroll
        for (int ni = 0; ni < 4; ++ni)
            #pragma unroll
            for (int r = 0; r < 8; ++r) {
                int gm = m0 + wm + mi * 16 + r + 8 * half;
                int gn = n0 + wn + ni * 16 + lr;
                if (gm < M && gn < N) {
                    long idx = (long)(rowbase + gm) * ldc + gn;
                    float v = acc[mi][ni][r];
                    if (Dz) v += Dz[idx];
                    Cz[idx] = v;
                }
            }
}

// =====================================================================
// RMSNorm over a row of `width`
// =====================================================================
__global__ __launch_bounds__(256)
void rmsnorm_kernel(const float* __restrict__ in, const float* __restrict__ w,
                    float* __restrict__ out, int width)
{
    long t = blockIdx.x;
    const float* x = in + t * width;
    float* o = out + t * width;
    __shared__ float red[256];
    float s = 0.0f;
    for (int j = threadIdx.x; j < width; j += 256) { float v = x[j]; s += v * v; }
    red[threadIdx.x] = s; __syncthreads();
    for (int o2 = 128; o2 > 0; o2 >>= 1) {
        if (threadIdx.x < (unsigned)o2) red[threadIdx.x] += red[threadIdx.x + o2];
        __syncthreads();
    }
    float inv = rsqrtf(red[0] / (float)width + EPSc);
    for (int j = threadIdx.x; j < width; j += 256) o[j] = x[j] * inv * w[j];
}

// =====================================================================
// Per-head RMSNorm + RoPE + GQA repeat.
// role y: [0,NH)          -> q head y    -> qrep[b][y][s][d]
//         [NH,NH+NKV)     -> k head      -> krep[b][kv*4 + r][s][d]  (4 replicas)
//         [NH+NKV,NH+2NKV)-> v copy      -> vrep (4 replicas)
// =====================================================================
__global__ __launch_bounds__(128)
void qknorm_rope_repeat(const float* __restrict__ qtmp, const float* __restrict__ ktmp,
                        const float* __restrict__ vtmp,
                        const float* __restrict__ qw, const float* __restrict__ kw,
                        const float* __restrict__ cosb, const float* __restrict__ sinb,
                        float* __restrict__ qrep, float* __restrict__ krep,
                        float* __restrict__ vrep)
{
    int t = blockIdx.x;
    int role = blockIdx.y;
    int d = threadIdx.x;
    int b = t / Sc, s = t % Sc;
    __shared__ float red[128];

    if (role < NHc + NKVc) {
        bool isq = role < NHc;
        int head = isq ? role : (role - NHc);
        const float* src = isq ? (qtmp + (long)t * (NHc * Dc) + head * Dc)
                               : (ktmp + (long)t * (NKVc * Dc) + head * Dc);
        const float* wn = isq ? qw : kw;
        float x = src[d];
        red[d] = x * x; __syncthreads();
        for (int o = 64; o > 0; o >>= 1) {
            if (d < o) red[d] += red[d + o];
            __syncthreads();
        }
        float inv = rsqrtf(red[0] / (float)Dc + EPSc);
        float xn = x * inv * wn[d];
        int pd = (d < 64) ? (d + 64) : (d - 64);
        float xr = src[pd] * inv * wn[pd];
        if (d < 64) xr = -xr;
        float out = xn * cosb[(long)s * Dc + d] + xr * sinb[(long)s * Dc + d];
        if (isq) {
            qrep[(((long)(b * NHc + head)) * Sc + s) * Dc + d] = out;
        } else {
            #pragma unroll
            for (int r = 0; r < NHc / NKVc; ++r)
                krep[(((long)(b * NHc + head * (NHc / NKVc) + r)) * Sc + s) * Dc + d] = out;
        }
    } else {
        int kv = role - NHc - NKVc;
        float x = vtmp[(long)t * (NKVc * Dc) + kv * Dc + d];
        #pragma unroll
        for (int r = 0; r < NHc / NKVc; ++r)
            vrep[(((long)(b * NHc + kv * (NHc / NKVc) + r)) * Sc + s) * Dc + d] = x;
    }
}

// =====================================================================
// Causal softmax row-wise (applies 1/sqrt(D) scale, zeroes j > i)
// =====================================================================
__global__ __launch_bounds__(256)
void softmax_causal(float* __restrict__ scores)
{
    int i = blockIdx.x;
    int z = blockIdx.y;
    float* row = scores + ((long)z * Sc + i) * Sc;
    int tid = threadIdx.x;
    const float scale = 0.088388347648318447f;   // 1/sqrt(128)
    __shared__ float red[256];

    float m = -1e30f;
    for (int j = tid; j <= i; j += 256) m = fmaxf(m, row[j] * scale);
    red[tid] = m; __syncthreads();
    for (int o = 128; o > 0; o >>= 1) {
        if (tid < o) red[tid] = fmaxf(red[tid], red[tid + o]);
        __syncthreads();
    }
    m = red[0]; __syncthreads();

    float ssum = 0.0f;
    for (int j = tid; j <= i; j += 256) {
        float p = __expf(row[j] * scale - m);
        row[j] = p; ssum += p;
    }
    red[tid] = ssum; __syncthreads();
    for (int o = 128; o > 0; o >>= 1) {
        if (tid < o) red[tid] += red[tid + o];
        __syncthreads();
    }
    float inv = 1.0f / red[0];
    for (int j = tid; j <= i; j += 256) row[j] *= inv;
    for (int j = i + 1 + tid; j < Sc; j += 256) row[j] = 0.0f;
}

// =====================================================================
// Router: logits = x @ gate_w^T, softmax(E=16), top-2, renormalize.
// =====================================================================
__global__ __launch_bounds__(256)
void router_kernel(const float* __restrict__ x, const float* __restrict__ gw,
                   int* __restrict__ counts, int* __restrict__ topk_id,
                   float* __restrict__ topk_w)
{
    int t = blockIdx.x, tid = threadIdx.x;
    __shared__ float red[256];
    __shared__ float logit[Ec];
    for (int e = 0; e < Ec; ++e) {
        float p = 0.0f;
        for (int j = tid; j < Hc; j += 256)
            p += x[(long)t * Hc + j] * gw[(long)e * Hc + j];
        red[tid] = p; __syncthreads();
        for (int o = 128; o > 0; o >>= 1) {
            if (tid < o) red[tid] += red[tid + o];
            __syncthreads();
        }
        if (tid == 0) logit[e] = red[0];
        __syncthreads();
    }
    if (tid == 0) {
        float mx = -1e30f;
        for (int e = 0; e < Ec; ++e) mx = fmaxf(mx, logit[e]);
        float p[Ec], s = 0.0f;
        for (int e = 0; e < Ec; ++e) { p[e] = __expf(logit[e] - mx); s += p[e]; }
        for (int e = 0; e < Ec; ++e) p[e] /= s;
        int i0 = 0;
        for (int e = 1; e < Ec; ++e) if (p[e] > p[i0]) i0 = e;
        int i1 = (i0 == 0) ? 1 : 0;
        for (int e = 0; e < Ec; ++e) if (e != i0 && p[e] > p[i1]) i1 = e;
        float w0 = p[i0], w1 = p[i1], sw = w0 + w1;
        topk_id[t * 2]     = i0;  topk_w[t * 2]     = w0 / sw;
        topk_id[t * 2 + 1] = i1;  topk_w[t * 2 + 1] = w1 / sw;
        atomicAdd(&counts[i0], 1);
        atomicAdd(&counts[i1], 1);
    }
}

__global__ void scan16_kernel(const int* __restrict__ counts,
                              int* __restrict__ offsets, int* __restrict__ cursor)
{
    if (blockIdx.x == 0 && threadIdx.x == 0) {
        int a = 0;
        for (int e = 0; e < Ec; ++e) { offsets[e] = a; cursor[e] = a; a += counts[e]; }
    }
}

__global__ __launch_bounds__(256)
void build_lists(const int* __restrict__ topk_id, int* __restrict__ cursor,
                 int* __restrict__ pair_token, int* __restrict__ slot_of)
{
    int t = blockIdx.x * blockDim.x + threadIdx.x;
    if (t >= Tc) return;
    for (int kk = 0; kk < Kc; ++kk) {
        int e = topk_id[t * 2 + kk];
        int slot = atomicAdd(&cursor[e], 1);
        pair_token[slot] = t;
        slot_of[t * 2 + kk] = slot;
    }
}

// =====================================================================
// h = silu(g) * u for gu = [g | u] rows of width 2*Fw
// =====================================================================
__global__ __launch_bounds__(256)
void silu_mul(const float* __restrict__ gu, float* __restrict__ h, int rows, int Fw)
{
    long idx = (long)blockIdx.x * blockDim.x + threadIdx.x;
    long total = (long)rows * Fw;
    if (idx >= total) return;
    long r = idx / Fw, c = idx % Fw;
    float g = gu[r * (2L * Fw) + c];
    float u = gu[r * (2L * Fw) + Fw + c];
    h[idx] = (g / (1.0f + __expf(-g))) * u;
}

// =====================================================================
// hidden[t] = shared[t] (already in out) + RSF * sum_k w_k * y[slot_k]
// =====================================================================
__global__ __launch_bounds__(256)
void combine_kernel(float* __restrict__ out, const float* __restrict__ y,
                    const int* __restrict__ slot_of, const float* __restrict__ topk_w)
{
    int t = blockIdx.x, tid = threadIdx.x;
    int s0 = slot_of[t * 2], s1 = slot_of[t * 2 + 1];
    float w0 = topk_w[t * 2], w1 = topk_w[t * 2 + 1];
    for (int hh = tid; hh < Hc; hh += 256) {
        float moe = w0 * y[(long)s0 * Hc + hh] + w1 * y[(long)s1 * Hc + hh];
        out[(long)t * Hc + hh] += RSFc * moe;
    }
}

// =====================================================================
// host: orchestration
// =====================================================================
static inline void launch_gemm(hipStream_t st,
                               const float* A, const float* B, const float* Cadd, float* C,
                               int M, int N, int Kd,
                               long lda, long ldb, long ldc,
                               long sA1, long sA2, long sB1, long sB2, long sC1, long sC2,
                               int zdiv, int Z, int transB,
                               const int* cnt, const int* off, const int* gather,
                               int gridM_override = 0)
{
    int gm = gridM_override ? gridM_override : (M + BM - 1) / BM;
    dim3 grid((N + BN - 1) / BN, gm, Z);
    gemm_bf16<<<grid, 256, 0, st>>>(A, B, Cadd, C, M, N, Kd, lda, ldb, ldc,
                                    sA1, sA2, sB1, sB2, sC1, sC2, zdiv, transB,
                                    cnt, off, gather);
}

extern "C" void kernel_launch(void* const* d_in, const int* in_sizes, int n_in,
                              void* d_out, int out_size, void* d_ws, size_t ws_size,
                              hipStream_t stream)
{
    (void)in_sizes; (void)n_in; (void)out_size; (void)ws_size;
    // inputs (setup_inputs order)
    const float* hidden = (const float*)d_in[0];
    const float* cosb   = (const float*)d_in[1];
    const float* sinb   = (const float*)d_in[2];
    const float* ln1_w  = (const float*)d_in[3];
    const float* ln2_w  = (const float*)d_in[4];
    const float* qnw    = (const float*)d_in[5];
    const float* knw    = (const float*)d_in[6];
    const float* wq     = (const float*)d_in[7];
    const float* wk     = (const float*)d_in[8];
    const float* wv     = (const float*)d_in[9];
    const float* wo     = (const float*)d_in[10];
    const float* gate_w = (const float*)d_in[11];
    const float* egu    = (const float*)d_in[12];
    const float* edn    = (const float*)d_in[13];
    const float* sgu_w  = (const float*)d_in[14];
    const float* sdn_w  = (const float*)d_in[15];

    float* out_hidden = (float*)d_out;                       // (B,S,H)
    float* out_resid  = (float*)d_out + (long)Tc * Hc;       // (B,S,H)

    // ---- workspace carve ----
    char* wp = (char*)d_ws;
    auto carve = [&](size_t bytes) -> void* {
        void* r = (void*)wp;
        wp += (bytes + 255) & ~(size_t)255;
        return r;
    };
    float* xn    = (float*)carve((size_t)Tc * Hc * 4);             // normed1
    float* qtmp  = (float*)carve((size_t)Tc * NHc * Dc * 4);
    float* ktmp  = (float*)carve((size_t)Tc * NKVc * Dc * 4);
    float* vtmp  = (float*)carve((size_t)Tc * NKVc * Dc * 4);
    float* qrep  = (float*)carve((size_t)Bc * NHc * Sc * Dc * 4);  // [b][h][s][d]
    float* krep  = (float*)carve((size_t)Bc * NHc * Sc * Dc * 4);
    float* vrep  = (float*)carve((size_t)Bc * NHc * Sc * Dc * 4);
    float* scores= (float*)carve((size_t)Bc * NHc * Sc * Sc * 4);  // [bh][s][s]
    float* obuf  = (float*)carve((size_t)Tc * NHc * Dc * 4);       // [b][s][h*d]
    float* x2n   = (float*)carve((size_t)Tc * Hc * 4);             // normed2
    float* sgu   = (float*)carve((size_t)Tc * 2 * FSc * 4);
    float* shg   = (float*)carve((size_t)Tc * FSc * 4);
    float* gu    = (float*)carve((size_t)(Tc * Kc) * 2 * Fc * 4);  // 4096 pair rows
    float* hb    = (float*)carve((size_t)(Tc * Kc) * Fc * 4);
    float* yb    = (float*)carve((size_t)(Tc * Kc) * Hc * 4);
    int*   counts    = (int*)carve(Ec * 4);
    int*   offsets   = (int*)carve(Ec * 4);
    int*   cursor    = (int*)carve(Ec * 4);
    int*   topk_id   = (int*)carve((size_t)Tc * 2 * 4);
    float* topk_w    = (float*)carve((size_t)Tc * 2 * 4);
    int*   pair_tok  = (int*)carve((size_t)Tc * Kc * 4);
    int*   slot_of   = (int*)carve((size_t)Tc * 2 * 4);

    const long SD  = (long)Sc * Dc;
    const long SS  = (long)Sc * Sc;

    // 1) ln1
    rmsnorm_kernel<<<Tc, 256, 0, stream>>>(hidden, ln1_w, xn, Hc);

    // 2) QKV projections (WMMA)
    launch_gemm(stream, xn, wq, nullptr, qtmp, Tc, NHc * Dc,  Hc, Hc, NHc * Dc,  NHc * Dc,
                0, 0, 0, 0, 0, 0, 1, 1, 0, nullptr, nullptr, nullptr);
    launch_gemm(stream, xn, wk, nullptr, ktmp, Tc, NKVc * Dc, Hc, Hc, NKVc * Dc, NKVc * Dc,
                0, 0, 0, 0, 0, 0, 1, 1, 0, nullptr, nullptr, nullptr);
    launch_gemm(stream, xn, wv, nullptr, vtmp, Tc, NKVc * Dc, Hc, Hc, NKVc * Dc, NKVc * Dc,
                0, 0, 0, 0, 0, 0, 1, 1, 0, nullptr, nullptr, nullptr);

    // 3) per-head rmsnorm + rope + GQA repeat
    qknorm_rope_repeat<<<dim3(Tc, NHc + 2 * NKVc), 128, 0, stream>>>(
        qtmp, ktmp, vtmp, qnw, knw, cosb, sinb, qrep, krep, vrep);

    // 4) scores = Q @ K^T, batched over z = b*NH + h (WMMA, transB)
    launch_gemm(stream, qrep, krep, nullptr, scores, Sc, Sc, Dc,
                Dc, Dc, Sc,
                (long)NHc * SD, SD, (long)NHc * SD, SD, (long)NHc * SS, SS,
                NHc, Bc * NHc, 1, nullptr, nullptr, nullptr);

    // 5) causal softmax (scale applied here)
    softmax_causal<<<dim3(Sc, Bc * NHc), 256, 0, stream>>>(scores);

    // 6) O = P @ V -> obuf[b][s][h*d] (WMMA)
    launch_gemm(stream, scores, vrep, nullptr, obuf, Sc, Dc, Sc,
                Sc, Dc, (long)NHc * Dc,
                (long)NHc * SS, SS, (long)NHc * SD, SD, (long)Sc * NHc * Dc, (long)Dc,
                NHc, Bc * NHc, 0, nullptr, nullptr, nullptr);

    // 7) attn_out @ wo + residual -> out_resid (WMMA, fused add)
    launch_gemm(stream, obuf, wo, hidden, out_resid, Tc, Hc, NHc * Dc,
                NHc * Dc, Hc, Hc, 0, 0, 0, 0, 0, 0, 1, 1, 0, nullptr, nullptr, nullptr);

    // 8) ln2 on residual
    rmsnorm_kernel<<<Tc, 256, 0, stream>>>(out_resid, ln2_w, x2n, Hc);

    // 9) routing: counts -> scan -> compact per-expert token lists
    hipMemsetAsync(counts, 0, Ec * sizeof(int), stream);
    router_kernel<<<Tc, 256, 0, stream>>>(x2n, gate_w, counts, topk_id, topk_w);
    scan16_kernel<<<1, 32, 0, stream>>>(counts, offsets, cursor);
    build_lists<<<(Tc + 255) / 256, 256, 0, stream>>>(topk_id, cursor, pair_tok, slot_of);

    // 10) expert gate_up (gathered rows, z = expert) (WMMA)
    launch_gemm(stream, x2n, egu, nullptr, gu, Tc, 2 * Fc, Hc,
                Hc, 2 * Fc, 2 * Fc,
                0, 0, (long)Hc * 2 * Fc, 0, 0, 0,
                1, Ec, 0, counts, offsets, pair_tok, (Tc + BM - 1) / BM);

    // 11) silu*mul on all token-expert pair rows
    {
        long total = (long)(Tc * Kc) * Fc;
        silu_mul<<<(unsigned)((total + 255) / 256), 256, 0, stream>>>(gu, hb, Tc * Kc, Fc);
    }

    // 12) expert down (contiguous pair-row ranges, z = expert) (WMMA)
    launch_gemm(stream, hb, edn, nullptr, yb, Tc, Hc, Fc,
                Fc, Hc, Hc,
                0, 0, (long)Fc * Hc, 0, 0, 0,
                1, Ec, 0, counts, offsets, nullptr, (Tc + BM - 1) / BM);

    // 13) shared expert: gate_up -> silu*mul -> down (WMMA), writes hidden base
    launch_gemm(stream, x2n, sgu_w, nullptr, sgu, Tc, 2 * FSc, Hc,
                Hc, 2 * FSc, 2 * FSc, 0, 0, 0, 0, 0, 0, 1, 1, 0, nullptr, nullptr, nullptr);
    {
        long total = (long)Tc * FSc;
        silu_mul<<<(unsigned)((total + 255) / 256), 256, 0, stream>>>(sgu, shg, Tc, FSc);
    }
    launch_gemm(stream, shg, sdn_w, nullptr, out_hidden, Tc, Hc, FSc,
                FSc, Hc, Hc, 0, 0, 0, 0, 0, 0, 1, 1, 0, nullptr, nullptr, nullptr);

    // 14) combine: hidden = shared + RSF * weighted expert outputs
    combine_kernel<<<Tc, 256, 0, stream>>>(out_hidden, yb, slot_of, topk_w);
}
